// VisionAttention_69801808494816
// MI455X (gfx1250) — matrix operand order repaired and checked by reading, not compiled, not run on hardware
//
#include <hip/hip_runtime.h>
#include <stdint.h>

// ---------- types ----------
typedef __bf16 bf16_t;
typedef __attribute__((ext_vector_type(16))) __bf16 v16bf;
typedef __attribute__((ext_vector_type(8)))  __bf16 v8bf;
typedef __attribute__((ext_vector_type(4)))  __bf16 v4bf;
typedef __attribute__((ext_vector_type(8)))  float  v8f;
typedef __attribute__((ext_vector_type(4)))  float  v4f;

#define DIM 1024
#define NHEAD 16
#define HDIM 64
#define BATCH 4
#define SEQ 1024
#define NTOK (BATCH * SEQ)      // 4096
#define QKVC (3 * DIM)          // 3072

// ---------- scalar conversions ----------
__device__ __forceinline__ bf16_t f2bf(float f) {
    unsigned u = __builtin_bit_cast(unsigned, f);
    unsigned short b = (unsigned short)((u + 0x7FFFu + ((u >> 16) & 1u)) >> 16);
    return __builtin_bit_cast(bf16_t, b);
}
__device__ __forceinline__ float bf2f(bf16_t x) {
    unsigned short u = __builtin_bit_cast(unsigned short, x);
    return __builtin_bit_cast(float, (unsigned)u << 16);
}

// ---------- WMMA wrapper: D = A(16x32 bf16) * B(32x16 bf16) + C(f32) ----------
__device__ __forceinline__ v8f wmma_bf16(v16bf a, v16bf b, v8f c) {
    return __builtin_amdgcn_wmma_f32_16x16x32_bf16(
        /*neg_a=*/false, a, /*neg_b=*/false, b,
        /*c_mod=*/(short)0, c, /*reuse_a=*/false, /*reuse_b=*/false);
}

// ---------- fragment loaders (pure bf16, 16B vector loads) ----------
// A 16x32 bf16 frag: lane<16 -> row=lane, K = 0..15; lane>=16 -> K = 16..31
__device__ __forceinline__ v16bf ldA_bf16(const bf16_t* base, int stride, int r0, int k0) {
    int lane = threadIdx.x & 31;
    int hw = lane >> 4, r = lane & 15;
    const bf16_t* p = base + (size_t)(r0 + r) * stride + k0 + hw * 16;
    union { v16bf v; v8bf h[2]; } u;
    u.h[0] = *(const v8bf*)(p);
    u.h[1] = *(const v8bf*)(p + 8);
    return u.v;
}
// B 32x16 bf16 frag: lane<16 -> col=lane, K = 0..15; lane>=16 -> K = 16..31.
// Source must be column-contiguous: base row c holds column c of B.
__device__ __forceinline__ v16bf ldB_bf16(const bf16_t* base, int stride, int c0, int k0) {
    int lane = threadIdx.x & 31;
    int hw = lane >> 4, c = lane & 15;
    const bf16_t* p = base + (size_t)(c0 + c) * stride + k0 + hw * 16;
    union { v16bf v; v8bf h[2]; } u;
    u.h[0] = *(const v8bf*)(p);
    u.h[1] = *(const v8bf*)(p + 8);
    return u.v;
}

// ============================================================================
// Kernel 0: bulk fp32 -> bf16 conversion (one pass; removes all conversion
// VALU work from the GEMM hot loops). 4 elements / thread, b128 in, b64 out.
// ============================================================================
__global__ void cvt_f32_bf16_kernel(const float* __restrict__ src,
                                    bf16_t* __restrict__ dst, int n4) {
    int i = blockIdx.x * blockDim.x + threadIdx.x;
    if (i >= n4) return;
    v4f f = *(const v4f*)(src + (size_t)i * 4);
    v4bf o;
    o[0] = f2bf(f[0]); o[1] = f2bf(f[1]); o[2] = f2bf(f[2]); o[3] = f2bf(f[3]);
    *(v4bf*)(dst + (size_t)i * 4) = o;
}

// ============================================================================
// Kernel 1: QKV GEMM (bf16 in, bf16 WMMA, fp32 acc) + bias; scatter into
// Q,K : [B,H,N,D] bf16 (pre-RoPE), Vt : [B,H,D,N] bf16 (transposed).
// Block = 128 threads (4 waves); wave tile 32x32 -> block tile 64x64.
// Per K-step: 2 A-frags + 2 B-frags feed 4 WMMAs (2x operand reuse).
// ============================================================================
__global__ void qkv_gemm_kernel(const bf16_t* __restrict__ xb,   // [NTOK, C] bf16
                                const bf16_t* __restrict__ wb,   // [3C, C]  bf16
                                const float* __restrict__ bias,  // [3C]
                                bf16_t* __restrict__ Qb,
                                bf16_t* __restrict__ Kb,
                                bf16_t* __restrict__ Vt) {
    int lane = threadIdx.x & 31;
    int wid  = threadIdx.x >> 5;
    int m0 = blockIdx.y * 64 + (wid >> 1) * 32;
    int n0 = blockIdx.x * 64 + (wid & 1) * 32;

    v8f acc00 = {}, acc01 = {}, acc10 = {}, acc11 = {};
    for (int k0 = 0; k0 < DIM; k0 += 32) {
        if (k0 + 32 < DIM) {   // hint next K-chunk (global_prefetch_b8)
            __builtin_prefetch(xb + (size_t)(m0 + (lane & 15)) * DIM + k0 + 32, 0, 1);
            __builtin_prefetch(wb + (size_t)(n0 + (lane & 15)) * DIM + k0 + 32, 0, 1);
        }
        v16bf a0 = ldA_bf16(xb, DIM, m0,      k0);
        v16bf a1 = ldA_bf16(xb, DIM, m0 + 16, k0);
        v16bf b0 = ldB_bf16(wb, DIM, n0,      k0);
        v16bf b1 = ldB_bf16(wb, DIM, n0 + 16, k0);
        acc00 = wmma_bf16(a0, b0, acc00);
        acc01 = wmma_bf16(a0, b1, acc01);
        acc10 = wmma_bf16(a1, b0, acc10);
        acc11 = wmma_bf16(a1, b1, acc11);
    }

    int hw = lane >> 4, col = lane & 15;
    v8f* accs[4] = {&acc00, &acc01, &acc10, &acc11};
#pragma unroll
    for (int tm = 0; tm < 2; ++tm) {
#pragma unroll
        for (int tn = 0; tn < 2; ++tn) {
            v8f& acc = *accs[tm * 2 + tn];
            int c = n0 + tn * 16 + col;
            float bv = bias[c];
            int t = c >> 10, rem = c & 1023, h = rem >> 6, d = rem & 63;
#pragma unroll
            for (int j = 0; j < 8; ++j) {
                int m = m0 + tm * 16 + hw * 8 + j;
                int bb = m >> 10, ntok = m & 1023;
                float v = acc[j] + bv;
                if (t == 2) {
                    Vt[(((size_t)(bb * NHEAD + h) * HDIM) + d) * SEQ + ntok] = f2bf(v);
                } else {
                    bf16_t* dst = t ? Kb : Qb;
                    dst[(((size_t)(bb * NHEAD + h) * SEQ) + ntok) * HDIM + d] = f2bf(v);
                }
            }
        }
    }
}

// ============================================================================
// Kernel 2: RoPE in place on Q,K. Pair (d, d+32) shares angle (ang is
// concat([ang,ang])). Folds attention scale 1/sqrt(D)=0.125 into Q.
// ============================================================================
__global__ void rope_kernel(bf16_t* __restrict__ Qb, bf16_t* __restrict__ Kb,
                            const float* __restrict__ cosb,   // [N, 64]
                            const float* __restrict__ sinb) { // [N, 64]
    int tid = blockIdx.x * blockDim.x + threadIdx.x;          // 0 .. 2*B*H*N*32-1
    const int PER = BATCH * NHEAD * SEQ * 32;                  // 2097152
    int which = tid >= PER;
    int rem = which ? tid - PER : tid;
    int bhn = rem >> 5;
    int d = rem & 31;
    int n = bhn & (SEQ - 1);
    bf16_t* base = (which ? Kb : Qb) + (size_t)bhn * HDIM;
    float lo = bf2f(base[d]);
    float hi = bf2f(base[d + 32]);
    float c = cosb[n * HDIM + d];
    float s = sinb[n * HDIM + d];
    float olo = lo * c - hi * s;
    float ohi = hi * c + lo * s;
    if (!which) { olo *= 0.125f; ohi *= 0.125f; }
    base[d]      = f2bf(olo);
    base[d + 32] = f2bf(ohi);
}

// ============================================================================
// Kernel 3: flash attention. One wave owns a 16-query block of one (b,h),
// streams keys in blocks of 32. Per step: 4 WMMAs for S, online softmax
// (shfl-xor row reductions within 16-lane half-groups), bf16 P through a
// 1 KB per-wave LDS tile (C-layout -> A-layout transpose), 4 WMMAs for PV.
// Output: ctx bf16 [NTOK, C] for the projection GEMM.
// ============================================================================
__global__ void flash_attn_kernel(const bf16_t* __restrict__ Qb,
                                  const bf16_t* __restrict__ Kb,
                                  const bf16_t* __restrict__ Vt,
                                  bf16_t* __restrict__ ctx) {
    __shared__ __align__(16) bf16_t Pbuf[4][16 * 32];   // 4 KB, one tile per wave

    int lane = threadIdx.x & 31;
    int wid  = threadIdx.x >> 5;
    int gw   = blockIdx.x * 4 + wid;     // global wave id: 0 .. 4095
    int bh   = gw >> 6;                  // (b*16+h), 0..63
    int q0   = (gw & 63) * 16;           // query block base in sequence

    const bf16_t* Qh  = Qb + (size_t)bh * SEQ * HDIM;
    const bf16_t* Kh  = Kb + (size_t)bh * SEQ * HDIM;
    const bf16_t* Vth = Vt + (size_t)bh * HDIM * SEQ;
    bf16_t* Pw = &Pbuf[wid][0];

    int hw = lane >> 4, col = lane & 15;

    // Q fragments (loaded once): d-chunks 0..31 and 32..63
    v16bf qa0 = ldA_bf16(Qh, HDIM, q0, 0);
    v16bf qa1 = ldA_bf16(Qh, HDIM, q0, 32);

    v8f accd[4] = {v8f{}, v8f{}, v8f{}, v8f{}};   // out 16 x 64 (4 d-tiles)
    float mrow[8], lrow[8];
#pragma unroll
    for (int j = 0; j < 8; ++j) { mrow[j] = -3.0e38f; lrow[j] = 0.0f; }

    for (int kb = 0; kb < SEQ; kb += 32) {
        // S = Q @ K^T : two 16x16 f32 tiles (keys kb..kb+15, kb+16..kb+31)
        v8f s0 = {}, s1 = {};
        s0 = wmma_bf16(qa0, ldB_bf16(Kh, HDIM, kb,      0),  s0);
        s0 = wmma_bf16(qa1, ldB_bf16(Kh, HDIM, kb,      32), s0);
        s1 = wmma_bf16(qa0, ldB_bf16(Kh, HDIM, kb + 16, 0),  s1);
        s1 = wmma_bf16(qa1, ldB_bf16(Kh, HDIM, kb + 16, 32), s1);

        // online softmax per C-fragment row (row = hw*8 + j, cols across 16 lanes)
#pragma unroll
        for (int j = 0; j < 8; ++j) {
            float cmax = fmaxf(s0[j], s1[j]);
#pragma unroll
            for (int off = 8; off >= 1; off >>= 1)
                cmax = fmaxf(cmax, __shfl_xor(cmax, off, 32));
            float mn = fmaxf(mrow[j], cmax);
            float alpha = __expf(mrow[j] - mn);
            float p0 = __expf(s0[j] - mn);
            float p1 = __expf(s1[j] - mn);
            float rs = p0 + p1;
#pragma unroll
            for (int off = 8; off >= 1; off >>= 1)
                rs += __shfl_xor(rs, off, 32);
            lrow[j] = lrow[j] * alpha + rs;
            mrow[j] = mn;
#pragma unroll
            for (int t = 0; t < 4; ++t) accd[t][j] *= alpha;
            int row = hw * 8 + j;
            Pw[row * 32 + col]      = f2bf(p0);
            Pw[row * 32 + 16 + col] = f2bf(p1);
        }

        // wave-local LDS turnaround: stores -> loads (CDNA5 split DS counter)
        asm volatile("s_wait_dscnt 0x0" ::: "memory");

        // P (A-layout from LDS) @ V (B from transposed Vt, contiguous keys)
        v16bf pa = ldA_bf16(Pw, 32, 0, 0);
#pragma unroll
        for (int t = 0; t < 4; ++t) {
            v16bf vb = ldB_bf16(Vth, SEQ, t * 16, kb);
            accd[t] = wmma_bf16(pa, vb, accd[t]);
        }
    }

    // normalize and emit ctx [token, h*64+d] bf16
    int b = bh >> 4, h = bh & 15;
#pragma unroll
    for (int j = 0; j < 8; ++j) {
        float inv = 1.0f / lrow[j];
        int ntok = q0 + hw * 8 + j;
        size_t trow = ((size_t)b * SEQ + ntok) * DIM;
#pragma unroll
        for (int t = 0; t < 4; ++t) {
            int ch = h * HDIM + t * 16 + col;
            ctx[trow + ch] = f2bf(accd[t][j] * inv);
        }
    }
}

// ============================================================================
// Kernel 4: projection GEMM: out[m,n] = ctx[m,:] . proj_w[n,:] + proj_b[n]
// Same 32x32-per-wave tiling; all-bf16 operand loads; fp32 out.
// ============================================================================
__global__ void proj_gemm_kernel(const bf16_t* __restrict__ ctx,
                                 const bf16_t* __restrict__ wb,   // [C, C] bf16
                                 const float* __restrict__ bias,  // [C]
                                 float* __restrict__ out) {
    int lane = threadIdx.x & 31;
    int wid  = threadIdx.x >> 5;
    int m0 = blockIdx.y * 64 + (wid >> 1) * 32;
    int n0 = blockIdx.x * 64 + (wid & 1) * 32;

    v8f acc00 = {}, acc01 = {}, acc10 = {}, acc11 = {};
    for (int k0 = 0; k0 < DIM; k0 += 32) {
        if (k0 + 32 < DIM) {
            __builtin_prefetch(ctx + (size_t)(m0 + (lane & 15)) * DIM + k0 + 32, 0, 1);
            __builtin_prefetch(wb  + (size_t)(n0 + (lane & 15)) * DIM + k0 + 32, 0, 1);
        }
        v16bf a0 = ldA_bf16(ctx, DIM, m0,      k0);
        v16bf a1 = ldA_bf16(ctx, DIM, m0 + 16, k0);
        v16bf b0 = ldB_bf16(wb,  DIM, n0,      k0);
        v16bf b1 = ldB_bf16(wb,  DIM, n0 + 16, k0);
        acc00 = wmma_bf16(a0, b0, acc00);
        acc01 = wmma_bf16(a0, b1, acc01);
        acc10 = wmma_bf16(a1, b0, acc10);
        acc11 = wmma_bf16(a1, b1, acc11);
    }

    int hw = lane >> 4, col = lane & 15;
    v8f* accs[4] = {&acc00, &acc01, &acc10, &acc11};
#pragma unroll
    for (int tm = 0; tm < 2; ++tm) {
#pragma unroll
        for (int tn = 0; tn < 2; ++tn) {
            v8f& acc = *accs[tm * 2 + tn];
            int n = n0 + tn * 16 + col;
            float bv = bias[n];
#pragma unroll
            for (int j = 0; j < 8; ++j) {
                int m = m0 + tm * 16 + hw * 8 + j;
                out[(size_t)m * DIM + n] = acc[j] + bv;
            }
        }
    }
}

// ============================================================================
// Host launcher
// Inputs (setup_inputs order): x, cos, sin, qkv_w, qkv_b, proj_w, proj_b (all f32)
// Output: fp32 [B, N, C] = [4096, 1024]
// Workspace (bf16): Q 8M | K 8M | Vt 8M | ctx 8M | x 8M | qkv_w 6M | proj_w 2M
// ============================================================================
extern "C" void kernel_launch(void* const* d_in, const int* in_sizes, int n_in,
                              void* d_out, int out_size, void* d_ws, size_t ws_size,
                              hipStream_t stream) {
    (void)in_sizes; (void)n_in; (void)out_size; (void)ws_size;
    const float* x      = (const float*)d_in[0];
    const float* cosb   = (const float*)d_in[1];
    const float* sinb   = (const float*)d_in[2];
    const float* qkv_w  = (const float*)d_in[3];
    const float* qkv_b  = (const float*)d_in[4];
    const float* proj_w = (const float*)d_in[5];
    const float* proj_b = (const float*)d_in[6];
    float* out = (float*)d_out;

    const size_t TEN   = (size_t)BATCH * NHEAD * SEQ * HDIM * sizeof(bf16_t); // 8 MB
    const size_t XSZ   = (size_t)NTOK * DIM * sizeof(bf16_t);                  // 8 MB
    const size_t WQSZ  = (size_t)QKVC * DIM * sizeof(bf16_t);                  // 6 MB
    uint8_t* ws = (uint8_t*)d_ws;
    bf16_t* Qb   = (bf16_t*)(ws);
    bf16_t* Kb   = (bf16_t*)(ws + TEN);
    bf16_t* Vt   = (bf16_t*)(ws + 2 * TEN);
    bf16_t* ctx  = (bf16_t*)(ws + 3 * TEN);
    bf16_t* xb   = (bf16_t*)(ws + 4 * TEN);
    bf16_t* wqkv = (bf16_t*)(ws + 4 * TEN + XSZ);
    bf16_t* wprj = (bf16_t*)(ws + 4 * TEN + XSZ + WQSZ);

    // 0) one-shot fp32 -> bf16 for GEMM operands (keeps K-loops VALU-free)
    cvt_f32_bf16_kernel<<<(NTOK * DIM / 4 + 255) / 256, 256, 0, stream>>>(x, xb, NTOK * DIM / 4);
    cvt_f32_bf16_kernel<<<(QKVC * DIM / 4 + 255) / 256, 256, 0, stream>>>(qkv_w, wqkv, QKVC * DIM / 4);
    cvt_f32_bf16_kernel<<<(DIM * DIM / 4 + 255) / 256, 256, 0, stream>>>(proj_w, wprj, DIM * DIM / 4);

    // 1) QKV GEMM + bias -> Q,K ([B,H,N,D]) and Vt ([B,H,D,N])
    qkv_gemm_kernel<<<dim3(QKVC / 64, NTOK / 64), 128, 0, stream>>>(
        xb, wqkv, qkv_b, Qb, Kb, Vt);

    // 2) RoPE in place on Q (with 1/sqrt(D) folded in) and K
    {
        int total = 2 * BATCH * NHEAD * SEQ * 32;   // 4,194,304
        rope_kernel<<<total / 256, 256, 0, stream>>>(Qb, Kb, cosb, sinb);
    }

    // 3) flash attention -> ctx bf16 [NTOK, C]
    flash_attn_kernel<<<(BATCH * NHEAD * (SEQ / 16)) / 4, 128, 0, stream>>>(
        Qb, Kb, Vt, ctx);

    // 4) projection GEMM + bias -> fp32 out
    proj_gemm_kernel<<<dim3(DIM / 64, NTOK / 64), 128, 0, stream>>>(
        ctx, wprj, proj_b, out);
}
